// UpConv_24739011625523
// MI455X (gfx1250) — matrix operand order repaired and checked by reading, not compiled, not run on hardware
//
#include <hip/hip_runtime.h>
#include <hip/hip_bf16.h>

// ---------------------------------------------------------------------------
// Shapes: B=4, CIN=128, COUT=64, HUP=128, RESO=256, NPTS=100000
// ---------------------------------------------------------------------------
#define NB 4
#define NPTS 100000
#define HP 258          // 256 + 2 halo

typedef __bf16 bf16;
typedef bf16 bf16x16 __attribute__((ext_vector_type(16)));
typedef float floatx8 __attribute__((ext_vector_type(8)));

union AFrag { bf16x16 v; bf16 e[16]; uint4 q[2]; };
union DFrag { floatx8 v; float e[8]; };

static __device__ inline floatx8 fzero() {
  DFrag d;
#pragma unroll
  for (int i = 0; i < 8; ++i) d.e[i] = 0.f;
  return d.v;
}

// 16-bit A fragment (16x32, MxK), row-major source A[m][k] = base[m*lda + k].
// ISA layout: lanes 0-15 (M=lane) hold K 0..7 (v0..3) and 16..23 (v4..7);
// lanes 16-31 (M=lane-16) hold K 8..15 and 24..31.  -> two b128 loads/lane.
static __device__ inline bf16x16 load_a_bf16(const bf16* base, int lda, int lane) {
  const bf16* pp = base + (lane & 15) * lda + ((lane >> 4) << 3);
  AFrag f;
  f.q[0] = *reinterpret_cast<const uint4*>(pp);
  f.q[1] = *reinterpret_cast<const uint4*>(pp + 16);
  return f.v;
}

// B fragment from pre-packed weights: block of 512 bf16 laid out [lane][16].
static __device__ inline bf16x16 load_b_packed(const bf16* block, int lane) {
  const bf16* pp = block + lane * 16;
  AFrag f;
  f.q[0] = *reinterpret_cast<const uint4*>(pp);
  f.q[1] = *reinterpret_cast<const uint4*>(pp + 8);
  return f.v;
}

// A fragment from fp32 source (converted to bf16 on the fly).
static __device__ inline bf16x16 load_a_f32(const float* base, int lda, int lane) {
  const float* pp = base + (lane & 15) * lda + ((lane >> 4) << 3);
  AFrag f;
#pragma unroll
  for (int i = 0; i < 8; ++i) { f.e[i] = (bf16)pp[i]; f.e[8 + i] = (bf16)pp[16 + i]; }
  return f.v;
}

static __device__ inline floatx8 wmma_bf16f32(bf16x16 a, bf16x16 b, floatx8 c) {
  return __builtin_amdgcn_wmma_f32_16x16x32_bf16(false, a, false, b, (short)0, c,
                                                 false, false);
}

// ---------------------------------------------------------------------------
// Utility kernels
// ---------------------------------------------------------------------------
__global__ __launch_bounds__(256) void k_zero(uint4* p, long n) {
  long i = (long)blockIdx.x * blockDim.x + threadIdx.x;
  long st = (long)gridDim.x * blockDim.x;
  uint4 z; z.x = z.y = z.z = z.w = 0u;
  for (; i < n; i += st) p[i] = z;
}

// Pack a weight tensor into B-fragment blocks: dst[((t*kcs+kc)*nts+nt)*512 + lane*16 + i]
// = W[o][ic][t]  (io=0: OIHW)  or  W[ic][o][t]  (io=1: IOHW)
// where o = nt*16 + (lane&15), ic = kc*32 + (lane>>4)*8 + (i&7) + (i>>3)*16.
__global__ __launch_bounds__(256) void k_pack(const float* __restrict__ W,
                                              bf16* __restrict__ dst,
                                              int O, int I, int T, int io, int total) {
  int tid = blockIdx.x * 256 + threadIdx.x;
  if (tid >= total) return;
  int i = tid & 15;
  int lane = (tid >> 4) & 31;
  int blk = tid >> 9;
  int nts = O >> 4, kcs = I >> 5;
  int nt = blk % nts; blk /= nts;
  int kc = blk % kcs; blk /= kcs;
  int t = blk;
  int o = nt * 16 + (lane & 15);
  int ic = kc * 32 + ((lane >> 4) << 3) + (i & 7) + ((i >> 3) << 4);
  float w = io ? W[(ic * O + o) * T + t] : W[(o * I + ic) * T + t];
  dst[tid] = (bf16)w;
}

// NCHW f32 -> NHWC bf16 (no halo), C=H=W=128 here.
__global__ __launch_bounds__(256) void k_nchw2nhwc(const float* __restrict__ src,
                                                   bf16* __restrict__ dst,
                                                   int C, int H, int W, int total) {
  int tid = blockIdx.x * 256 + threadIdx.x;
  if (tid >= total) return;
  int c = tid % C; int r = tid / C;
  int w = r % W; r /= W;
  int h = r % H; int b = r / H;
  dst[tid] = (bf16)src[((b * C + c) * H + h) * W + w];
}

// from_down (B,64,256,256) f32 NCHW -> xcat channels 64..127, interior of halo buf.
__global__ __launch_bounds__(256) void k_filldown(const float* __restrict__ src,
                                                  bf16* __restrict__ xcat, int total) {
  int tid = blockIdx.x * 256 + threadIdx.x;
  if (tid >= total) return;
  int c = tid & 63; int t = tid >> 6;
  int x = t & 255; t >>= 8;
  int y = t & 255; int b = t >> 8;
  xcat[(((b * HP + y + 1) * HP) + x + 1) * 128 + 64 + c] =
      (bf16)src[((b * 64 + c) * 256 + y) * 256 + x];
}

// ---------------------------------------------------------------------------
// Transposed conv 2x2 stride 2 (WMMA).  Input NHWC bf16 (B,128,128,128).
// Wave = 16 input pixels x 16 out channels.  All 16 B fragments (4 sub-pixel
// taps x 4 K-chunks) are loop-invariant -> hoisted into registers so the
// steady-state loop is one A load + 4 back-to-back WMMAs.
// ---------------------------------------------------------------------------
__global__ __launch_bounds__(128) void k_upconv1(const bf16* __restrict__ xin,
                                                 const bf16* __restrict__ wp,
                                                 const float* __restrict__ bias,
                                                 bf16* __restrict__ xcat) {
  int wave = threadIdx.x >> 5, lane = threadIdx.x & 31;
  int bid = blockIdx.x;                 // NB*128*8
  int wt = bid & 7; int t = bid >> 3;
  int h = t & 127; int b = t >> 7;
  int w0 = wt * 16;

  bf16x16 bw[16];
#pragma unroll
  for (int dk = 0; dk < 4; ++dk)
#pragma unroll
    for (int kc = 0; kc < 4; ++kc)
      bw[dk * 4 + kc] = load_b_packed(wp + ((dk * 4 + kc) * 4 + wave) * 512, lane);

  floatx8 acc[4];
#pragma unroll
  for (int dk = 0; dk < 4; ++dk) acc[dk] = fzero();
  const bf16* ab = xin + (((b * 128 + h) * 128) + w0) * 128;
#pragma unroll
  for (int kc = 0; kc < 4; ++kc) {
    bf16x16 a = load_a_bf16(ab + kc * 32, 128, lane);
#pragma unroll
    for (int dk = 0; dk < 4; ++dk)
      acc[dk] = wmma_bf16f32(a, bw[dk * 4 + kc], acc[dk]);
  }

  int o = wave * 16 + (lane & 15);
  float bv = bias[o];
  int mb = (lane >> 4) << 3;
#pragma unroll
  for (int dk = 0; dk < 4; ++dk) {
    int d = dk >> 1, k = dk & 1;
    DFrag df; df.v = acc[dk];
#pragma unroll
    for (int r = 0; r < 8; ++r) {
      int m = mb + r;
      int yo = 1 + 2 * h + d;
      int xo = 1 + 2 * (w0 + m) + k;
      xcat[(((b * HP + yo) * HP) + xo) * 128 + o] = (bf16)(df.e[r] + bv);
    }
  }
}

// Same GEMM but accumulate-into fp32 NCHW x output (the skip upconv).
__global__ __launch_bounds__(128) void k_upconv_add(const bf16* __restrict__ xin,
                                                    const bf16* __restrict__ wp,
                                                    const float* __restrict__ bias,
                                                    float* __restrict__ xout) {
  int wave = threadIdx.x >> 5, lane = threadIdx.x & 31;
  int bid = blockIdx.x;
  int wt = bid & 7; int t = bid >> 3;
  int h = t & 127; int b = t >> 7;
  int w0 = wt * 16;

  bf16x16 bw[16];
#pragma unroll
  for (int dk = 0; dk < 4; ++dk)
#pragma unroll
    for (int kc = 0; kc < 4; ++kc)
      bw[dk * 4 + kc] = load_b_packed(wp + ((dk * 4 + kc) * 4 + wave) * 512, lane);

  floatx8 acc[4];
#pragma unroll
  for (int dk = 0; dk < 4; ++dk) acc[dk] = fzero();
  const bf16* ab = xin + (((b * 128 + h) * 128) + w0) * 128;
#pragma unroll
  for (int kc = 0; kc < 4; ++kc) {
    bf16x16 a = load_a_bf16(ab + kc * 32, 128, lane);
#pragma unroll
    for (int dk = 0; dk < 4; ++dk)
      acc[dk] = wmma_bf16f32(a, bw[dk * 4 + kc], acc[dk]);
  }

  int o = wave * 16 + (lane & 15);
  float bv = bias[o];
  int mb = (lane >> 4) << 3;
#pragma unroll
  for (int dk = 0; dk < 4; ++dk) {
    int d = dk >> 1, k = dk & 1;
    DFrag df; df.v = acc[dk];
#pragma unroll
    for (int r = 0; r < 8; ++r) {
      int m = mb + r;
      int yo = 2 * h + d;
      int xo = 2 * (w0 + m) + k;
      int idx = ((b * 64 + o) * 256 + yo) * 256 + xo;
      xout[idx] += df.e[r] + bv;
    }
  }
}

// ---------------------------------------------------------------------------
// Conv3x3 implicit GEMM (WMMA), LDS-staged input tile + 2x M blocking.
// Block = 32 output pixels (one y row) x 64 out channels; 4 waves = 4 n-tiles;
// each wave holds 2 accumulators (two 16-pixel M tiles sharing each B frag).
// Input tile (3 rows x 34 cols x CI, halo incl.) staged once into LDS, so all
// A fragments come from ds_load_b128 and the 4 waves share one copy.
// PADOUT=true: ReLU -> padded NHWC bf16 (C=64). PADOUT=false: ReLU -> NCHW f32.
// ---------------------------------------------------------------------------
template <int CI, bool PADOUT>
__global__ __launch_bounds__(128) void k_conv3x3(const bf16* __restrict__ xin,
                                                 const bf16* __restrict__ wp,
                                                 const float* __restrict__ bias,
                                                 bf16* __restrict__ outp,
                                                 float* __restrict__ outn) {
  constexpr int KCS = CI / 32;
  constexpr int TILEW = 34;                 // 32 pixels + 2 halo columns
  constexpr int ROWE = TILEW * CI;          // elems per staged row
  __shared__ __align__(16) bf16 tile[3 * ROWE];

  int tid = threadIdx.x;
  int wave = tid >> 5, lane = tid & 31;
  int bid = blockIdx.x;                     // NB*256*8
  int xt = bid & 7; int t = bid >> 3;
  int y = t & 255; int b = t >> 8;
  int x0 = xt * 32;

  // ---- cooperative stage: 3 contiguous row segments of the padded input ----
#pragma unroll
  for (int rr = 0; rr < 3; ++rr) {
    const bf16* src = xin + (((b * HP + y + rr) * HP) + x0) * CI;
    bf16* dst = tile + rr * ROWE;
    for (int e = tid * 8; e < ROWE; e += 128 * 8)
      *reinterpret_cast<uint4*>(dst + e) =
          *reinterpret_cast<const uint4*>(src + e);
  }
  __syncthreads();

  floatx8 acc0 = fzero(), acc1 = fzero();
  for (int ty = 0; ty < 3; ++ty) {
#pragma unroll
    for (int tx = 0; tx < 3; ++tx) {
      const bf16* ab = tile + (ty * TILEW + tx) * CI;
      int tap = ty * 3 + tx;
      __builtin_prefetch(wp + (((tap + 1) * KCS) * 4 + wave) * 512, 0, 1);
#pragma unroll
      for (int kc = 0; kc < KCS; ++kc) {
        bf16x16 a0 = load_a_bf16(ab + kc * 32, CI, lane);
        bf16x16 a1 = load_a_bf16(ab + 16 * CI + kc * 32, CI, lane);
        bf16x16 bb = load_b_packed(wp + ((tap * KCS + kc) * 4 + wave) * 512, lane);
        acc0 = wmma_bf16f32(a0, bb, acc0);
        acc1 = wmma_bf16f32(a1, bb, acc1);
      }
    }
  }

  int o = wave * 16 + (lane & 15);
  float bv = bias[o];
  int mb = (lane >> 4) << 3;
  DFrag d0; d0.v = acc0;
  DFrag d1; d1.v = acc1;
#pragma unroll
  for (int r = 0; r < 8; ++r) {
    float v0 = fmaxf(d0.e[r] + bv, 0.f);
    float v1 = fmaxf(d1.e[r] + bv, 0.f);
    int xx0 = x0 + mb + r;
    int xx1 = xx0 + 16;
    if constexpr (PADOUT) {
      outp[(((b * HP + y + 1) * HP) + xx0 + 1) * 64 + o] = (bf16)v0;
      outp[(((b * HP + y + 1) * HP) + xx1 + 1) * 64 + o] = (bf16)v1;
    } else {
      outn[((b * 64 + o) * 256 + y) * 256 + xx0] = v0;
      outn[((b * 64 + o) * 256 + y) * 256 + xx1] = v1;
    }
  }
}

// ---------------------------------------------------------------------------
// Bilinear sampling of x (NCHW f32) at 400k points; also compute scatter index.
// ---------------------------------------------------------------------------
__global__ __launch_bounds__(256) void k_sample(const float* __restrict__ p,
                                                const float* __restrict__ x,
                                                bf16* __restrict__ csamp,
                                                int* __restrict__ gidx, int total) {
  int tid = blockIdx.x * 256 + threadIdx.x;
  if (tid >= total) return;
  int ch = tid & 63; int t = tid >> 6;
  int pt = t % NPTS; int b = t / NPTS;
  const float* pp = p + (b * NPTS + pt) * 3;
  float px = pp[0], py = pp[1];
  float ix = fminf(fmaxf(px * 255.f, 0.f), 255.f);
  float iy = fminf(fmaxf(py * 255.f, 0.f), 255.f);
  float x0f = floorf(ix), y0f = floorf(iy);
  int x0 = (int)x0f, y0 = (int)y0f;
  int x1 = min(x0 + 1, 255), y1 = min(y0 + 1, 255);
  float wx = ix - x0f, wy = iy - y0f;
  const float* f = x + (b * 64 + ch) * 65536;
  float f00 = f[y0 * 256 + x0], f01 = f[y0 * 256 + x1];
  float f10 = f[y1 * 256 + x0], f11 = f[y1 * 256 + x1];
  float top = f00 * (1.f - wx) + f01 * wx;
  float bot = f10 * (1.f - wx) + f11 * wx;
  csamp[tid] = (bf16)(top * (1.f - wy) + bot * wy);
  if (ch == 0) {
    int gx = min(max((int)(px * 256.f), 0), 255);
    int gy = min(max((int)(py * 256.f), 0), 255);
    gidx[b * NPTS + pt] = gx + 256 * gy;
  }
}

// ---------------------------------------------------------------------------
// Per-point MLP (WMMA): relu(c@Wf1^T+bf1)@Wf2^T + bf2 + c_last@Wfc^T + bfc.
// Wave = 16 points.  Hidden (16x128) staged through LDS to re-form A frags.
// ---------------------------------------------------------------------------
__global__ __launch_bounds__(128) void k_mlp(const bf16* __restrict__ csamp,
                                             const float* __restrict__ clast,
                                             const bf16* __restrict__ wf1p,
                                             const bf16* __restrict__ wf2p,
                                             const bf16* __restrict__ wfcp,
                                             const float* __restrict__ bf1,
                                             const float* __restrict__ bf2,
                                             const float* __restrict__ bfc,
                                             float* __restrict__ cout) {
  __shared__ __align__(16) bf16 hbuf[4][16 * 128];
  int wave = threadIdx.x >> 5, lane = threadIdx.x & 31;
  int tile = blockIdx.x * 4 + wave;     // 25000 tiles = NB * (NPTS/16)
  int b = tile / (NPTS / 16);
  int p0 = (tile % (NPTS / 16)) * 16;
  int nlo = lane & 15, mb = (lane >> 4) << 3;

  const bf16* ab = csamp + (b * NPTS + p0) * 64;
  bf16x16 a0 = load_a_bf16(ab, 64, lane);
  bf16x16 a1 = load_a_bf16(ab + 32, 64, lane);
#pragma unroll
  for (int nt = 0; nt < 8; ++nt) {
    floatx8 acc = fzero();
    acc = wmma_bf16f32(a0, load_b_packed(wf1p + (0 * 8 + nt) * 512, lane), acc);
    acc = wmma_bf16f32(a1, load_b_packed(wf1p + (1 * 8 + nt) * 512, lane), acc);
    int n = nt * 16 + nlo;
    float bv = bf1[n];
    DFrag df; df.v = acc;
#pragma unroll
    for (int r = 0; r < 8; ++r)
      hbuf[wave][(mb + r) * 128 + n] = (bf16)fmaxf(df.e[r] + bv, 0.f);
  }
  __syncthreads();

  const float* cl = clast + (b * NPTS + p0) * 128;
#pragma unroll
  for (int nt = 0; nt < 4; ++nt) {
    floatx8 acc = fzero();
#pragma unroll
    for (int kc = 0; kc < 4; ++kc) {
      bf16x16 ah = load_a_bf16(&hbuf[wave][0] + kc * 32, 128, lane);
      acc = wmma_bf16f32(ah, load_b_packed(wf2p + (kc * 4 + nt) * 512, lane), acc);
      bf16x16 ac = load_a_f32(cl + kc * 32, 128, lane);
      acc = wmma_bf16f32(ac, load_b_packed(wfcp + (kc * 4 + nt) * 512, lane), acc);
    }
    int o = nt * 16 + nlo;
    float bv = bf2[o] + bfc[o];
    DFrag df; df.v = acc;
#pragma unroll
    for (int r = 0; r < 8; ++r)
      cout[(b * NPTS + p0 + mb + r) * 64 + o] = df.e[r] + bv;
  }
}

// ---------------------------------------------------------------------------
// Scatter-mean: atomic sums + counts, then normalize into fea (NCHW).
// ---------------------------------------------------------------------------
__global__ __launch_bounds__(256) void k_scatter(const float* __restrict__ c,
                                                 const int* __restrict__ gidx,
                                                 float* __restrict__ sums,
                                                 float* __restrict__ cnts, int total) {
  int tid = blockIdx.x * 256 + threadIdx.x;
  if (tid >= total) return;
  int ch = tid & 63; int t = tid >> 6;
  int pt = t % NPTS; int b = t / NPTS;
  int g = gidx[b * NPTS + pt];
  atomicAdd(&sums[(b * 65536 + g) * 64 + ch], c[tid]);
  if (ch == 0) atomicAdd(&cnts[b * 65536 + g], 1.0f);
}

__global__ __launch_bounds__(256) void k_finalize(const float* __restrict__ sums,
                                                  const float* __restrict__ cnts,
                                                  float* __restrict__ fea, int total) {
  int tid = blockIdx.x * 256 + threadIdx.x;
  if (tid >= total) return;
  int g = tid & 65535; int t = tid >> 16;
  int ch = t & 63; int b = t >> 6;
  float cnt = fmaxf(cnts[b * 65536 + g], 1.0f);
  fea[tid] = sums[(b * 65536 + g) * 64 + ch] / cnt;
}

// ---------------------------------------------------------------------------
// Launcher
// ---------------------------------------------------------------------------
extern "C" void kernel_launch(void* const* d_in, const int* in_sizes, int n_in,
                              void* d_out, int out_size, void* d_ws, size_t ws_size,
                              hipStream_t stream) {
  (void)in_sizes; (void)n_in; (void)out_size; (void)ws_size;
  const float* p         = (const float*)d_in[0];
  const float* from_up   = (const float*)d_in[1];
  const float* from_down = (const float*)d_in[2];
  const float* x_ac      = (const float*)d_in[3];
  const float* c_last    = (const float*)d_in[4];
  const float* W_up = (const float*)d_in[5];  const float* b_up = (const float*)d_in[6];
  const float* W1   = (const float*)d_in[7];  const float* b1   = (const float*)d_in[8];
  const float* W2   = (const float*)d_in[9];  const float* b2   = (const float*)d_in[10];
  const float* Wc   = (const float*)d_in[11]; const float* bc   = (const float*)d_in[12];
  const float* Wf1  = (const float*)d_in[13]; const float* bf1  = (const float*)d_in[14];
  const float* Wf2  = (const float*)d_in[15]; const float* bf2  = (const float*)d_in[16];
  const float* Wfc  = (const float*)d_in[17]; const float* bfc  = (const float*)d_in[18];

  char* ws = (char*)d_ws;
  bf16*  xcat  = (bf16*)(ws + 0);            // (4,258,258,128) bf16  68,161,536 B
  bf16*  t1    = (bf16*)(ws + 68161536);     // (4,258,258,64)  bf16  34,080,768 B
  float* sums  = (float*)(ws + 102242304);   // (4,65536,64)    f32   67,108,864 B
  float* cnts  = (float*)(ws + 169351168);   // (4,65536)       f32    1,048,576 B
  bf16*  xup   = (bf16*)(ws + 170399744);    // (4,128,128,128) bf16  16,777,216 B
  bf16*  xac   = (bf16*)(ws + 187176960);    // (4,128,128,128) bf16  16,777,216 B
  bf16*  csamp = (bf16*)(ws + 203954176);    // (4,100000,64)   bf16  51,200,000 B
  int*   gidx  = (int*)(ws + 255154176);     // (4,100000)      i32    1,600,000 B
  bf16*  wpk   = (bf16*)(ws + 256754176);    // packed weights       ~401,408 B
  bf16* Wupp = wpk + 0;
  bf16* W1p  = wpk + 32768;
  bf16* W2p  = wpk + 106496;
  bf16* Wcp  = wpk + 143360;
  bf16* Wf1p = wpk + 176128;
  bf16* Wf2p = wpk + 184320;
  bf16* Wfcp = wpk + 192512;

  float* out_fea = (float*)d_out;
  float* out_x   = out_fea + 16777216;
  float* out_c   = out_fea + 33554432;

  // 1) zero halo buffers + scatter accumulators (first 170,399,744 bytes)
  k_zero<<<4096, 256, 0, stream>>>((uint4*)ws, 170399744L / 16);

  // 2) pack all weights into B-fragment order
  auto pack = [&](const float* W, bf16* dst, int O, int I, int T, int io) {
    int total = T * (I / 32) * (O / 16) * 512;
    k_pack<<<(total + 255) / 256, 256, 0, stream>>>(W, dst, O, I, T, io, total);
  };
  pack(W_up, Wupp, 64, 128, 4, 1);   // (CIN,COUT,2,2) -> IOHW
  pack(W1,   W1p,  64, 128, 9, 0);   // OIHW
  pack(W2,   W2p,  64, 64,  9, 0);
  pack(Wc,   Wcp,  64, 128, 4, 1);
  pack(Wf1,  Wf1p, 128, 64, 1, 0);
  pack(Wf2,  Wf2p, 64, 128, 1, 0);
  pack(Wfc,  Wfcp, 64, 128, 1, 0);

  // 3) layout transforms
  k_nchw2nhwc<<<32768, 256, 0, stream>>>(from_up, xup, 128, 128, 128, 8388608);
  k_nchw2nhwc<<<32768, 256, 0, stream>>>(x_ac,   xac, 128, 128, 128, 8388608);
  k_filldown<<<65536, 256, 0, stream>>>(from_down, xcat, 16777216);

  // 4) upconv2x2 -> xcat channels 0..63   (B frags register-resident)
  k_upconv1<<<4096, 128, 0, stream>>>(xup, Wupp, b_up, xcat);

  // 5) conv1 3x3 (128->64) + ReLU -> padded NHWC t1   (LDS-staged, 2x M-block)
  k_conv3x3<128, true><<<8192, 128, 0, stream>>>(xcat, W1p, b1, t1, (float*)nullptr);

  // 6) conv2 3x3 (64->64) + ReLU -> x output (NCHW f32)
  k_conv3x3<64, false><<<8192, 128, 0, stream>>>(t1, W2p, b2, (bf16*)nullptr, out_x);

  // 7) skip upconv2x2 added into x
  k_upconv_add<<<4096, 128, 0, stream>>>(xac, Wcp, bc, out_x);

  // 8) bilinear point sampling (+ grid index)
  k_sample<<<100000, 256, 0, stream>>>(p, out_x, csamp, gidx, 25600000);

  // 9) per-point MLP -> c output
  k_mlp<<<6250, 128, 0, stream>>>(csamp, c_last, Wf1p, Wf2p, Wfcp, bf1, bf2, bfc, out_c);

  // 10) scatter-mean -> fea output
  k_scatter<<<100000, 256, 0, stream>>>(out_c, gidx, sums, cnts, 25600000);
  k_finalize<<<65536, 256, 0, stream>>>(sums, cnts, out_fea, 16777216);
}
